// GATEncoder_59931973649018
// MI455X (gfx1250) — matrix operand order repaired and checked by reading, not compiled, not run on hardware
//
#include <hip/hip_runtime.h>
#include <hip/hip_bf16.h>
#include <hip/hip_fp16.h>

// ---------------------------------------------------------------------------
// CDNA5 (gfx1250) WMMA types
// ---------------------------------------------------------------------------
typedef __attribute__((ext_vector_type(16))) _Float16 v16h;
typedef __attribute__((ext_vector_type(8)))  float    v8f;
typedef unsigned int v4u __attribute__((ext_vector_type(4)));
typedef int          v8i __attribute__((ext_vector_type(8)));
typedef int          v4i __attribute__((ext_vector_type(4)));

union V16H { v16h v; _Float16 h[16]; unsigned int u[8]; };

static __device__ __forceinline__ v8f wmma_f16(v16h a, v16h b, v8f c) {
  // 8 args: (neg_a, A, neg_b, B, c_mod, C, reuse_a, reuse_b)
  return __builtin_amdgcn_wmma_f32_16x16x32_f16(false, a, false, b, (short)0, c,
                                                false, false);
}

#define NN   4096   // nodes
#define DD   256    // hidden
#define HH   4      // heads
#define DH   256    // per-head dim
#define QKS  1024   // H*DH row stride of q/k/v

// ---------------------------------------------------------------------------
// Tensor Data Mover: async 2D f16 tile load global -> LDS (CDNA5 TDM).
// D# per CDNA5 ISA: group0 {count/lds_addr/global_addr/type=2},
// group1 {data_size=2B, tensor_dim0/1, tile_dim0/1, dim0 stride}. Groups 2/3
// zero (2D tensor, tile_dim2=0 => unused). Tracked with TENSORcnt.
// ---------------------------------------------------------------------------
static __device__ __forceinline__ void tdm_load_2d_f16(
    unsigned lds_off, const _Float16* gptr, unsigned tensor_d0,
    unsigned tensor_d1, unsigned tile_d0, unsigned tile_d1,
    unsigned stride_d0) {
  unsigned long long ga = (unsigned long long)gptr;
  v4u g0;
  g0[0] = 1u;                                // count=1, user descriptor
  g0[1] = lds_off;                           // lds_addr (bytes)
  g0[2] = (unsigned)(ga & 0xFFFFFFFFu);      // global_addr[31:0]
  g0[3] = (unsigned)((ga >> 32) & 0x01FFFFFFu) | 0x80000000u;  // [56:32]|type=2
  v8i g1;
  g1[0] = (int)(1u << 16);                               // data_size=1 (2 B)
  g1[1] = (int)((tensor_d0 & 0xFFFFu) << 16);            // tensor_dim0 lo
  g1[2] = (int)((tensor_d0 >> 16) | ((tensor_d1 & 0xFFFFu) << 16));
  g1[3] = (int)((tensor_d1 >> 16) | ((tile_d0 & 0xFFFFu) << 16));
  g1[4] = (int)(tile_d1 & 0xFFFFu);                      // tile_dim1, dim2=0
  g1[5] = (int)stride_d0;                                // dim0 stride lo
  g1[6] = 0;
  g1[7] = 0;
  v4i z4 = {0, 0, 0, 0};
#if defined(__clang_major__) && (__clang_major__ >= 23)
  v8i z8 = {0, 0, 0, 0, 0, 0, 0, 0};
  __builtin_amdgcn_tensor_load_to_lds(g0, g1, z4, z4, z8, 0);
#else
  __builtin_amdgcn_tensor_load_to_lds(g0, g1, z4, z4, 0);
#endif
}

// ===========================================================================
// Generic fused GEMM:  C[M,N] = act( A[M,K] @ B[K,N] + bias[N] (+ A residual) )
// block = 256 threads (8 waves).  Block tile 128(M) x 64(N), K stepped by 32.
// ACT: 0=none 1=relu 2=elu.  RES: add A[m*K+n] (requires N==K).  HOUT: f16 out.
// ===========================================================================
template <int ACT, bool RES, bool HOUT>
__global__ __launch_bounds__(256) void gemm_kernel(
    const float* __restrict__ A, const float* __restrict__ B,
    const float* __restrict__ bias, float* __restrict__ C,
    _Float16* __restrict__ Ch, int M, int N, int K) {
  __shared__ _Float16 lds_a[128][32];  // [row][k]      8 KB
  __shared__ _Float16 lds_b[64][32];   // [n][k] (B^T)  4 KB

  const int tid  = threadIdx.x;
  const int w    = tid >> 5;
  const int lane = tid & 31;
  const int lr   = lane & 15;
  const int lh   = lane >> 4;
  const int row0 = blockIdx.y * 128;
  const int col0 = blockIdx.x * 64;

  v8f acc[4];
#pragma unroll
  for (int c = 0; c < 4; ++c)
#pragma unroll
    for (int j = 0; j < 8; ++j) acc[c][j] = 0.0f;

  for (int k0 = 0; k0 < K; k0 += 32) {
    __syncthreads();
    // ---- stage A tile 128x32 (f32 -> f16), float4 vectorized ----
#pragma unroll
    for (int i = 0; i < 4; ++i) {
      int idx = tid + i * 256;  // 0..1023 float4 chunks
      int r   = idx >> 3;
      int kp  = idx & 7;
      const float* ap = A + (size_t)(row0 + r) * K + k0 + 4 * kp;
      float4 f = *(const float4*)ap;
      if (k0 + 32 < K) __builtin_prefetch(ap + 32, 0, 3);  // next K tile
      _Float16* dst = &lds_a[r][4 * kp];
      dst[0] = (_Float16)f.x; dst[1] = (_Float16)f.y;
      dst[2] = (_Float16)f.z; dst[3] = (_Float16)f.w;
    }
    // ---- stage B tile 32x64 transposed into lds_b[n][k] ----
#pragma unroll
    for (int i = 0; i < 8; ++i) {
      int idx = tid + i * 256;  // 0..2047
      int kk  = idx >> 6;
      int n   = idx & 63;
      lds_b[n][kk] = (_Float16)B[(size_t)(k0 + kk) * N + col0 + n];
    }
    __syncthreads();

    // ---- A fragment (16x32): lane lr = row, K halves per ISA layout ----
    V16H a;
    const _Float16* ar = &lds_a[w * 16 + lr][8 * lh];
#pragma unroll
    for (int p = 0; p < 8; ++p)
      a.u[p] = *(const unsigned int*)(ar + 16 * (p >> 2) + 2 * (p & 3));

#pragma unroll
    for (int c = 0; c < 4; ++c) {
      V16H b;  // B fragment (32x16): lane lr = col, K halves per ISA layout (same as A)
      const _Float16* br = &lds_b[c * 16 + lr][8 * lh];
#pragma unroll
      for (int p = 0; p < 8; ++p)
        b.u[p] = *(const unsigned int*)(br + 16 * (p >> 2) + 2 * (p & 3));
      acc[c] = wmma_f16(a.v, b.v, acc[c]);
    }
  }

  // ---- epilogue: C layout = VGPR j -> row j+8*lh, lane lr -> col ----
#pragma unroll
  for (int c = 0; c < 4; ++c)
#pragma unroll
    for (int j = 0; j < 8; ++j) {
      int m = row0 + w * 16 + j + 8 * lh;
      int n = col0 + c * 16 + lr;
      float v = acc[c][j] + bias[n];
      if constexpr (RES) v += A[(size_t)m * K + n];  // residual (N==K)
      if constexpr (ACT == 1) v = fmaxf(v, 0.0f);
      if constexpr (ACT == 2) v = (v > 0.0f) ? v : (__expf(v) - 1.0f);
      if constexpr (HOUT) Ch[(size_t)m * N + n] = (_Float16)v;
      else                C [(size_t)m * N + n] = v;
    }
}

// ===========================================================================
// Flash attention over the complete graph (dense MHA), one head per block.y.
// block = 128 threads (4 waves) -> 64 query rows; key blocks of 32 streamed.
// K tiles: double-buffered TDM (tensor_load_to_lds) prefetched 1 block ahead.
// V tiles: cooperative transposed staging (TDM cannot transpose).
// ===========================================================================
__global__ __launch_bounds__(128) void attn_kernel(
    const _Float16* __restrict__ Q, const _Float16* __restrict__ K,
    const _Float16* __restrict__ V, float* __restrict__ OH) {
  __shared__ _Float16 lds_k[2][32][256];  // [buf][key][dh]  32 KB (TDM dest)
  __shared__ _Float16 lds_v[256][32];     // [dh][key]       16 KB
  __shared__ _Float16 lds_p[4][16][32];   // per-wave P       4 KB

  const int tid   = threadIdx.x;
  const int w     = tid >> 5;
  const int lane  = tid & 31;
  const int lr    = lane & 15;
  const int lh    = lane >> 4;
  const int head  = blockIdx.y;
  const int qrow0 = blockIdx.x * 64 + w * 16;

  // ---- resident Q fragments: 8 dh-chunks of 32 ----
  V16H qf[8];
  {
    const _Float16* qp = Q + (size_t)(qrow0 + lr) * QKS + head * DH;
#pragma unroll
    for (int c = 0; c < 8; ++c)
#pragma unroll
      for (int p = 0; p < 8; ++p) {
        int dh = c * 32 + 16 * (p >> 2) + 8 * lh + 2 * (p & 3);
        qf[c].u[p] = *(const unsigned int*)(qp + dh);
      }
  }

  v8f o[16];  // O accumulator: 16 dh-chunks x (16x16 f32)
#pragma unroll
  for (int c = 0; c < 16; ++c)
#pragma unroll
    for (int j = 0; j < 8; ++j) o[c][j] = 0.0f;
  float mrow[8], lrow[8];
#pragma unroll
  for (int j = 0; j < 8; ++j) { mrow[j] = -1.0e30f; lrow[j] = 0.0f; }

  // ---- issue TDM for first K tile into buffer 0 ----
  if (w == 0)
    tdm_load_2d_f16((unsigned)(unsigned long long)&lds_k[0][0][0],
                    K + (size_t)head * DH, DH, NN, DH, 32, QKS);

  for (int it = 0, kb = 0; kb < NN; kb += 32, ++it) {
    const int buf = it & 1;
    __syncthreads();  // previous compute done: lds_v free, next K buf free

    // ---- cooperative transposed staging of V block [dh][key] ----
    {
      const unsigned int* vg =
          (const unsigned int*)(V + (size_t)kb * QKS + head * DH);
      if (kb + 32 < NN) __builtin_prefetch(vg + 32 * 512, 0, 3);
#pragma unroll 4
      for (int i = 0; i < 32; ++i) {
        int idx = tid + 128 * i;  // 0..4095 uint (dh pairs)
        int key = idx >> 7;
        int dp  = idx & 127;
        union { unsigned int u; _Float16 h[2]; } cv; cv.u = vg[key * 512 + dp];
        lds_v[2 * dp][key]     = cv.h[0];
        lds_v[2 * dp + 1][key] = cv.h[1];
      }
    }

    // ---- wait current K tile, prefetch next via TDM ----
    if (w == 0) {
      __builtin_amdgcn_s_wait_tensorcnt(0);  // K(kb) landed in lds_k[buf]
      if (kb + 32 < NN)
        tdm_load_2d_f16((unsigned)(unsigned long long)&lds_k[buf ^ 1][0][0],
                        K + (size_t)(kb + 32) * QKS + head * DH,
                        DH, NN, DH, 32, QKS);
    }
    __syncthreads();

    // ---- S = Q K^T / sqrt(DH): two 16x16 tiles (keys 0-15, 16-31) ----
    v8f s0, s1;
#pragma unroll
    for (int j = 0; j < 8; ++j) { s0[j] = 0.0f; s1[j] = 0.0f; }
#pragma unroll
    for (int c = 0; c < 8; ++c) {
      V16H b0, b1;
      const _Float16* kr0 = &lds_k[buf][lr][c * 32 + 8 * lh];
      const _Float16* kr1 = &lds_k[buf][16 + lr][c * 32 + 8 * lh];
#pragma unroll
      for (int p = 0; p < 8; ++p) {
        b0.u[p] = *(const unsigned int*)(kr0 + 16 * (p >> 2) + 2 * (p & 3));
        b1.u[p] = *(const unsigned int*)(kr1 + 16 * (p >> 2) + 2 * (p & 3));
      }
      s0 = wmma_f16(qf[c].v, b0.v, s0);
      s1 = wmma_f16(qf[c].v, b1.v, s1);
    }

    // ---- online softmax (rows live across 16-lane halves) ----
    float pv0[8], pv1[8];
#pragma unroll
    for (int j = 0; j < 8; ++j) {
      float a0 = s0[j] * 0.0625f;  // 1/sqrt(256)
      float a1 = s1[j] * 0.0625f;
      float mx = fmaxf(a0, a1);
#pragma unroll
      for (int m = 1; m < 16; m <<= 1) mx = fmaxf(mx, __shfl_xor(mx, m, 16));
      float mn    = fmaxf(mrow[j], mx);
      float alpha = __expf(mrow[j] - mn);
      mrow[j]     = mn;
      float p0 = __expf(a0 - mn), p1 = __expf(a1 - mn);
      float rs = p0 + p1;
#pragma unroll
      for (int m = 1; m < 16; m <<= 1) rs += __shfl_xor(rs, m, 16);
      lrow[j] = lrow[j] * alpha + rs;
      pv0[j] = p0; pv1[j] = p1;
#pragma unroll
      for (int c = 0; c < 16; ++c) o[c][j] *= alpha;
    }

    // ---- C-layout P -> LDS -> A-fragment (per-wave patch, in-order DS) ----
#pragma unroll
    for (int j = 0; j < 8; ++j) {
      lds_p[w][j + 8 * lh][lr]      = (_Float16)pv0[j];
      lds_p[w][j + 8 * lh][16 + lr] = (_Float16)pv1[j];
    }
    V16H pa;
    {
      const _Float16* pr = &lds_p[w][lr][0];
#pragma unroll
      for (int p = 0; p < 8; ++p)
        pa.u[p] = *(const unsigned int*)(pr + 16 * (p >> 2) + 8 * lh + 2 * (p & 3));
    }

    // ---- O += P @ V over 16 dh-chunks ----
#pragma unroll
    for (int c = 0; c < 16; ++c) {
      V16H bv;
      const _Float16* vr = &lds_v[c * 16 + lr][8 * lh];
#pragma unroll
      for (int p = 0; p < 8; ++p)
        bv.u[p] = *(const unsigned int*)(vr + 16 * (p >> 2) + 2 * (p & 3));
      o[c] = wmma_f16(pa.v, bv.v, o[c]);
    }
  }

  // ---- normalize and write per-head output ----
  float* op = OH + (size_t)head * NN * DH;
#pragma unroll
  for (int c = 0; c < 16; ++c)
#pragma unroll
    for (int j = 0; j < 8; ++j) {
      int row = qrow0 + j + 8 * lh;
      int col = c * 16 + lr;
      op[(size_t)row * DH + col] = o[c][j] / lrow[j];
    }
}

// ===========================================================================
// LayerNorm: 1 wave per row (D=256 -> 8 elems/lane), 4 rows per block.
// ===========================================================================
__global__ __launch_bounds__(128) void ln_kernel(
    const float* __restrict__ X, const float* __restrict__ G,
    const float* __restrict__ Bt, float* __restrict__ Y) {
  const int w = threadIdx.x >> 5, lane = threadIdx.x & 31;
  const int row = blockIdx.x * 4 + w;
  const float* xr = X + (size_t)row * DD;
  float x[8], s = 0.f, s2 = 0.f;
#pragma unroll
  for (int i = 0; i < 8; ++i) {
    x[i] = xr[lane + 32 * i]; s += x[i]; s2 += x[i] * x[i];
  }
#pragma unroll
  for (int m = 16; m >= 1; m >>= 1) {
    s += __shfl_xor(s, m, 32); s2 += __shfl_xor(s2, m, 32);
  }
  float mean = s * (1.f / DD);
  float rstd = rsqrtf(s2 * (1.f / DD) - mean * mean + 1e-5f);
  float* yr = Y + (size_t)row * DD;
#pragma unroll
  for (int i = 0; i < 8; ++i) {
    int d = lane + 32 * i;
    yr[d] = (x[i] - mean) * rstd * G[d] + Bt[d];
  }
}

// Fused: x = mean_h(OH) + skip ; then LayerNorm.
__global__ __launch_bounds__(128) void merge_ln_kernel(
    const float* __restrict__ OH, const float* __restrict__ SK,
    const float* __restrict__ G, const float* __restrict__ Bt,
    float* __restrict__ Y) {
  const int w = threadIdx.x >> 5, lane = threadIdx.x & 31;
  const int row = blockIdx.x * 4 + w;
  const size_t hs = (size_t)NN * DD;  // head stride
  float x[8], s = 0.f, s2 = 0.f;
#pragma unroll
  for (int i = 0; i < 8; ++i) {
    size_t o = (size_t)row * DD + lane + 32 * i;
    float v = 0.25f * (OH[o] + OH[o + hs] + OH[o + 2 * hs] + OH[o + 3 * hs]) + SK[o];
    x[i] = v; s += v; s2 += v * v;
  }
#pragma unroll
  for (int m = 16; m >= 1; m >>= 1) {
    s += __shfl_xor(s, m, 32); s2 += __shfl_xor(s2, m, 32);
  }
  float mean = s * (1.f / DD);
  float rstd = rsqrtf(s2 * (1.f / DD) - mean * mean + 1e-5f);
  float* yr = Y + (size_t)row * DD;
#pragma unroll
  for (int i = 0; i < 8; ++i) {
    int d = lane + 32 * i;
    yr[d] = (x[i] - mean) * rstd * G[d] + Bt[d];
  }
}

// ===========================================================================
// Orchestration
// ===========================================================================
extern "C" void kernel_launch(void* const* d_in, const int* in_sizes, int n_in,
                              void* d_out, int out_size, void* d_ws,
                              size_t ws_size, hipStream_t stream) {
  (void)in_sizes; (void)n_in; (void)out_size; (void)ws_size;
  const float* x      = (const float*)d_in[0];
  const float* w_in   = (const float*)d_in[1];
  const float* b_in   = (const float*)d_in[2];
  const float* w_m1   = (const float*)d_in[3];
  const float* b_m1   = (const float*)d_in[4];
  const float* g_m1   = (const float*)d_in[5];
  const float* be_m1  = (const float*)d_in[6];
  const float* wq     = (const float*)d_in[7];
  const float* bq     = (const float*)d_in[8];
  const float* wk     = (const float*)d_in[9];
  const float* bk     = (const float*)d_in[10];
  const float* wv     = (const float*)d_in[11];
  const float* bv     = (const float*)d_in[12];
  const float* wskip  = (const float*)d_in[13];
  const float* bskip  = (const float*)d_in[14];
  const float* g_n1   = (const float*)d_in[15];
  const float* be_n1  = (const float*)d_in[16];
  const float* w_m2   = (const float*)d_in[17];
  const float* b_m2   = (const float*)d_in[18];
  const float* g_m2   = (const float*)d_in[19];
  const float* be_m2  = (const float*)d_in[20];
  const float* w_mean = (const float*)d_in[21];
  const float* b_mean = (const float*)d_in[22];
  float* out = (float*)d_out;

  // workspace layout (64 MB total)
  const size_t ND = (size_t)NN * DD;  // 1,048,576 floats
  float* ws   = (float*)d_ws;
  float* h1   = ws;            // 4 MB
  float* t    = h1 + ND;       // 4 MB (reused)
  float* h2   = t + ND;        // 4 MB
  float* skip = h2 + ND;       // 4 MB
  float* h3   = skip + ND;     // 4 MB
  float* h4   = h3 + ND;       // 4 MB
  float* oh   = h4 + ND;       // 16 MB  [H][N][DH]
  _Float16* qh = (_Float16*)(oh + 4 * ND);  // 8 MB each (f16)
  _Float16* kh = qh + (size_t)NN * QKS;
  _Float16* vh = kh + (size_t)NN * QKS;

  const dim3 blk256(256), blk128(128);
  const dim3 gD(DD / 64, NN / 128);     // N=256 GEMMs
  const dim3 gQ(QKS / 64, NN / 128);    // N=1024 GEMMs
  const dim3 gO(128 / 64, NN / 128);    // N=128 GEMM

  // 1) h1 = elu(x @ w_in + b_in + x)
  gemm_kernel<2, true, false><<<gD, blk256, 0, stream>>>(
      x, w_in, b_in, h1, nullptr, NN, DD, DD);
  // 2) h2 = LN(relu(h1 @ w_m1 + b_m1 + h1))
  gemm_kernel<1, true, false><<<gD, blk256, 0, stream>>>(
      h1, w_m1, b_m1, t, nullptr, NN, DD, DD);
  ln_kernel<<<NN / 4, blk128, 0, stream>>>(t, g_m1, be_m1, h2);
  // 3) q/k/v (f16) and skip
  gemm_kernel<0, false, true><<<gQ, blk256, 0, stream>>>(
      h2, wq, bq, nullptr, qh, NN, QKS, DD);
  gemm_kernel<0, false, true><<<gQ, blk256, 0, stream>>>(
      h2, wk, bk, nullptr, kh, NN, QKS, DD);
  gemm_kernel<0, false, true><<<gQ, blk256, 0, stream>>>(
      h2, wv, bv, nullptr, vh, NN, QKS, DD);
  gemm_kernel<0, false, false><<<gD, blk256, 0, stream>>>(
      h2, wskip, bskip, skip, nullptr, NN, DD, DD);
  // 4) flash attention per head (TDM-prefetched K tiles)
  attn_kernel<<<dim3(NN / 64, HH), blk128, 0, stream>>>(qh, kh, vh, oh);
  // 5) h3 = LN(mean_h(oh) + skip)
  merge_ln_kernel<<<NN / 4, blk128, 0, stream>>>(oh, skip, g_n1, be_n1, h3);
  // 6) h4 = LN(relu(h3 @ w_m2 + b_m2 + h3))
  gemm_kernel<1, true, false><<<gD, blk256, 0, stream>>>(
      h3, w_m2, b_m2, t, nullptr, NN, DD, DD);
  ln_kernel<<<NN / 4, blk128, 0, stream>>>(t, g_m2, be_m2, h4);
  // 7) out = h4 @ w_mean + b_mean
  gemm_kernel<0, false, false><<<gO, blk256, 0, stream>>>(
      h4, w_mean, b_mean, out, nullptr, NN, 128, DD);
}
